// PolicyMultipleMPGNN_44598940401789
// MI455X (gfx1250) — compile-verified, matrix operand order checked by reading, not run on hardware
//
#include <hip/hip_runtime.h>

typedef _Float16 h16v __attribute__((ext_vector_type(16)));
typedef _Float16 h8v  __attribute__((ext_vector_type(8)));
typedef float    f8v  __attribute__((ext_vector_type(8)));

// leaky_relu(v, 0.01) == max(v, 0.01*v)  (valid since slope in (0,1))
// lowers to a single dual-issuable v_max_f32 instead of v_cmp + v_cndmask.
__device__ __forceinline__ float lrelu(float v) { return fmaxf(v, 0.01f * v); }

// ---------------------------------------------------------------------------
// WMMA fragment loader: 16x32 f16 tile (A layout; also used for B from
// pre-transposed weights Wt[N][K]).  Per CDNA5 ISA 7.12.2:
//   lanes 0-15 : M=lane, K = k0+0..7 (v0-3) and k0+16..23 (v4-7)
//   lanes 16-31: M=lane-16, K = k0+8..15 and k0+24..31
// -> two contiguous 16-byte loads per lane.
// ---------------------------------------------------------------------------
__device__ __forceinline__ h16v load_frag(const _Float16* base, int stride,
                                          int mb, int k0, int lane) {
  const int m  = lane & 15;
  const int ko = (lane >> 4) << 3;                 // 0 or 8
  const _Float16* p = base + (size_t)(mb + m) * stride + k0 + ko;
  union { h16v v; h8v h[2]; } u;
  u.h[0] = *(const h8v*)(p);
  u.h[1] = *(const h8v*)(p + 16);
  return u.v;
}

// ---------------------------------------------------------------------------
// One hidden GEMM layer: LDS f16 [64 x K] @ Wt^T -> LDS f16 [64 x NOUT],
// bias + leaky-relu epilogue. 8 waves cover MT*(NOUT/16) 16x16 tiles.
// ---------------------------------------------------------------------------
template <int K, int NOUT, int MT>
__device__ __forceinline__ void lds_gemm(const _Float16* Ain, int sA,
                                         _Float16* Aout, int sO,
                                         const _Float16* __restrict__ Wt,
                                         const float* __restrict__ bias,
                                         int lane, int wave) {
  constexpr int NT = NOUT / 16;
  for (int t = wave; t < MT * NT; t += 8) {
    const int mt = t / NT, nt = t % NT;
    f8v c = {};
#pragma unroll
    for (int k0 = 0; k0 < K; k0 += 32) {
      h16v a = load_frag(Ain, sA, mt * 16, k0, lane);
      h16v b = load_frag(Wt, K, nt * 16, k0, lane);
      c = __builtin_amdgcn_wmma_f32_16x16x32_f16(false, a, false, b,
                                                 (short)0, c, false, false);
    }
    const int cn = lane & 15;
    const int rb = (lane >> 4) << 3;
    const float bv = bias[nt * 16 + cn];
#pragma unroll
    for (int i = 0; i < 8; ++i) {
      const float v = lrelu(c[i] + bv);
      Aout[(size_t)(mt * 16 + rb + i) * sO + nt * 16 + cn] = (_Float16)v;
    }
  }
}

// ---------------------------------------------------------------------------
// Weight convert: f32 [P][K][N] -> f16 [P][N][K] (WMMA-friendly transpose)
// ---------------------------------------------------------------------------
__global__ void convert_w_kernel(const float* __restrict__ in,
                                 _Float16* __restrict__ out,
                                 int P, int K, int N) {
  int idx = blockIdx.x * 256 + threadIdx.x;
  int total = P * K * N;
  if (idx >= total) return;
  int p = idx / (K * N);
  int r = idx - p * (K * N);
  int n = r / K;
  int k = r - n * K;
  out[idx] = (_Float16)in[((size_t)p * K + k) * N + n];
}

// ---------------------------------------------------------------------------
// Embedding: out[m][d] = lrelu(in[m][0:16] @ W[16][64] + b[d])
// ---------------------------------------------------------------------------
__global__ void embed_kernel(const float* __restrict__ in,
                             const float* __restrict__ w,
                             const float* __restrict__ b,
                             float* __restrict__ out, int M) {
  int idx = blockIdx.x * 256 + threadIdx.x;
  if (idx >= M * 64) return;
  int m = idx >> 6, d = idx & 63;
  float acc = b[d];
#pragma unroll
  for (int k = 0; k < 16; ++k) acc += in[m * 16 + k] * w[k * 64 + d];
  out[idx] = lrelu(acc);
}

// ---------------------------------------------------------------------------
// Edge MLP: gather [64 edges x 256] -> 128 -> 128 -> 64, fused residual
// e += e_new and atomic scatter-add into e_agg[dst].
// Indices (row, col, batch[row]) are staged in LDS first so the main gather
// issues independent, pipelineable global loads (no dependent index chains).
// ---------------------------------------------------------------------------
__global__ __launch_bounds__(256) void edge_mlp_kernel(
    const float* __restrict__ xh, float* __restrict__ e,
    const float* __restrict__ g, const int* __restrict__ erow,
    const int* __restrict__ ecol, const int* __restrict__ nbatch,
    const _Float16* __restrict__ w1t, const float* __restrict__ b1,
    const _Float16* __restrict__ w2t, const float* __restrict__ b2,
    const _Float16* __restrict__ w3t, const float* __restrict__ b3,
    float* __restrict__ e_agg, int E) {
  __shared__ _Float16 A0[64 * 272];   // layer-1 input; re-used as layer-3 input
  __shared__ _Float16 A1[64 * 144];
  __shared__ int sRow[64], sCol[64], sGb[64];
  const int tid = threadIdx.x;
  const int lane = tid & 31, wave = tid >> 5;
  const int e0 = blockIdx.x * 64;
  const int k = tid;  // 256 features per edge, one per thread

  __builtin_prefetch(w1t, 0, 1);
  __builtin_prefetch(w2t, 0, 1);

  if (tid < 64) {
    const int eg = e0 + tid;
    sRow[tid] = (eg < E) ? erow[eg] : 0;
  } else if (tid < 128) {
    const int eg = e0 + tid - 64;
    sCol[tid - 64] = (eg < E) ? ecol[eg] : 0;
  } else if (tid < 192) {
    const int eg = e0 + tid - 128;
    sGb[tid - 128] = (eg < E) ? nbatch[erow[eg]] : 0;
  }
  __syncthreads();

  for (int i = 0; i < 64; ++i) {
    const int eg = e0 + i;
    float v = 0.f;
    if (eg < E) {
      if (k < 64)       v = xh[(size_t)sRow[i] * 64 + k];
      else if (k < 128) v = xh[(size_t)sCol[i] * 64 + (k - 64)];
      else if (k < 192) v = e[(size_t)eg * 64 + (k - 128)];
      else              v = g[(size_t)sGb[i] * 64 + (k - 192)];
    }
    A0[i * 272 + k] = (_Float16)v;
  }
  __syncthreads();
  lds_gemm<256, 128, 4>(A0, 272, A1, 144, w1t, b1, lane, wave);
  __syncthreads();
  lds_gemm<128, 128, 4>(A1, 144, A0, 144, w2t, b2, lane, wave);
  __syncthreads();
  // layer 3: 128 -> 64, no activation; fused residual + scatter
  for (int t = wave; t < 16; t += 8) {
    const int mt = t >> 2, nt = t & 3;
    f8v c = {};
#pragma unroll
    for (int k0 = 0; k0 < 128; k0 += 32) {
      h16v a = load_frag(A0, 144, mt * 16, k0, lane);
      h16v b = load_frag(w3t, 128, nt * 16, k0, lane);
      c = __builtin_amdgcn_wmma_f32_16x16x32_f16(false, a, false, b,
                                                 (short)0, c, false, false);
    }
    const int cn = lane & 15, rb = (lane >> 4) << 3;
    const int d = nt * 16 + cn;
    const float bv = b3[d];
#pragma unroll
    for (int i = 0; i < 8; ++i) {
      const int el = mt * 16 + rb + i;      // edge index within tile
      const int eg = e0 + el;
      if (eg < E) {
        const float v = c[i] + bv;
        e[(size_t)eg * 64 + d] += v;
        atomicAdd(&e_agg[(size_t)sCol[el] * 64 + d], v);
      }
    }
  }
}

// ---------------------------------------------------------------------------
// Node MLP: [64 nodes x 192] -> 128 -> 128 -> 64, fused xh += x_new and
// atomic scatter-add into x_agg[batch].
// ---------------------------------------------------------------------------
__global__ __launch_bounds__(256) void node_mlp_kernel(
    float* __restrict__ xh, const float* __restrict__ e_agg,
    const float* __restrict__ g, const int* __restrict__ nbatch,
    const _Float16* __restrict__ w1t, const float* __restrict__ b1,
    const _Float16* __restrict__ w2t, const float* __restrict__ b2,
    const _Float16* __restrict__ w3t, const float* __restrict__ b3,
    float* __restrict__ x_agg, int N) {
  __shared__ _Float16 A0[64 * 208];
  __shared__ _Float16 A1[64 * 144];
  __shared__ int sGb[64];
  const int tid = threadIdx.x;
  const int lane = tid & 31, wave = tid >> 5;
  const int n0 = blockIdx.x * 64;
  const int k = tid;

  if (tid < 64) {
    const int n = n0 + tid;
    sGb[tid] = (n < N) ? nbatch[n] : 0;
  }
  __syncthreads();

  for (int i = 0; i < 64; ++i) {
    const int n = n0 + i;
    if (k < 192) {
      float v = 0.f;
      if (n < N) {
        if (k < 64)       v = xh[(size_t)n * 64 + k];
        else if (k < 128) v = e_agg[(size_t)n * 64 + (k - 64)];
        else              v = g[(size_t)sGb[i] * 64 + (k - 128)];
      }
      A0[i * 208 + k] = (_Float16)v;
    }
  }
  __syncthreads();
  lds_gemm<192, 128, 4>(A0, 208, A1, 144, w1t, b1, lane, wave);
  __syncthreads();
  lds_gemm<128, 128, 4>(A1, 144, A0, 144, w2t, b2, lane, wave);
  __syncthreads();
  for (int t = wave; t < 16; t += 8) {
    const int mt = t >> 2, nt = t & 3;
    f8v c = {};
#pragma unroll
    for (int k0 = 0; k0 < 128; k0 += 32) {
      h16v a = load_frag(A0, 144, mt * 16, k0, lane);
      h16v b = load_frag(w3t, 128, nt * 16, k0, lane);
      c = __builtin_amdgcn_wmma_f32_16x16x32_f16(false, a, false, b,
                                                 (short)0, c, false, false);
    }
    const int cn = lane & 15, rb = (lane >> 4) << 3;
    const int d = nt * 16 + cn;
    const float bv = b3[d];
#pragma unroll
    for (int i = 0; i < 8; ++i) {
      const int nl = mt * 16 + rb + i;
      const int n = n0 + nl;
      if (n < N) {
        const float v = c[i] + bv;
        xh[(size_t)n * 64 + d] += v;
        atomicAdd(&x_agg[(size_t)sGb[nl] * 64 + d], v);
      }
    }
  }
}

// ---------------------------------------------------------------------------
// Global MLP: one workgroup, [B x 128] -> 128 -> 128 -> 64, g += g_new.
// ---------------------------------------------------------------------------
__global__ __launch_bounds__(256) void global_mlp_kernel(
    float* __restrict__ g, const float* __restrict__ x_agg,
    const _Float16* __restrict__ w1t, const float* __restrict__ b1,
    const _Float16* __restrict__ w2t, const float* __restrict__ b2,
    const _Float16* __restrict__ w3t, const float* __restrict__ b3,
    int B) {
  __shared__ _Float16 A0[64 * 144];
  __shared__ _Float16 A1[64 * 144];
  const int tid = threadIdx.x;
  const int lane = tid & 31, wave = tid >> 5;
  const int k = tid;

  for (int i = 0; i < 64; ++i) {
    if (k < 128) {
      float v = 0.f;
      if (i < B)
        v = (k < 64) ? g[(size_t)i * 64 + k] : x_agg[(size_t)i * 64 + (k - 64)];
      A0[i * 144 + k] = (_Float16)v;
    }
  }
  __syncthreads();
  lds_gemm<128, 128, 4>(A0, 144, A1, 144, w1t, b1, lane, wave);
  __syncthreads();
  lds_gemm<128, 128, 4>(A1, 144, A0, 144, w2t, b2, lane, wave);
  __syncthreads();
  for (int t = wave; t < 16; t += 8) {
    const int mt = t >> 2, nt = t & 3;
    f8v c = {};
#pragma unroll
    for (int k0 = 0; k0 < 128; k0 += 32) {
      h16v a = load_frag(A0, 144, mt * 16, k0, lane);
      h16v b = load_frag(w3t, 128, nt * 16, k0, lane);
      c = __builtin_amdgcn_wmma_f32_16x16x32_f16(false, a, false, b,
                                                 (short)0, c, false, false);
    }
    const int cn = lane & 15, rb = (lane >> 4) << 3;
    const int d = nt * 16 + cn;
    const float bv = b3[d];
#pragma unroll
    for (int i = 0; i < 8; ++i) {
      const int gi = mt * 16 + rb + i;
      if (gi < B) g[(size_t)gi * 64 + d] += c[i] + bv;
    }
  }
}

// ---------------------------------------------------------------------------
// Decoders
// ---------------------------------------------------------------------------
__global__ __launch_bounds__(256) void edge_decode_kernel(
    const float* __restrict__ e, const float* __restrict__ dec_w,
    const float* __restrict__ dec_b, float* __restrict__ out, int E) {
  const int eidx = blockIdx.x * 8 + (threadIdx.x >> 5);
  const int lane = threadIdx.x & 31;
  if (eidx >= E) return;
  float acc = e[(size_t)eidx * 64 + lane] * dec_w[lane] +
              e[(size_t)eidx * 64 + 32 + lane] * dec_w[32 + lane];
#pragma unroll
  for (int off = 16; off > 0; off >>= 1) acc += __shfl_xor(acc, off, 32);
  if (lane == 0) out[eidx] = acc + dec_b[0];
}

__global__ void value_kernel(const float* __restrict__ g,
                             const float* __restrict__ vw1,
                             const float* __restrict__ vb1,
                             const float* __restrict__ vw2,
                             const float* __restrict__ vb2,
                             float* __restrict__ out, int B) {
  const int gi = threadIdx.x;
  if (gi >= B) return;
  float acc = vb2[0];
  for (int d = 0; d < 64; ++d) {
    float h = vb1[d];
#pragma unroll
    for (int kk = 0; kk < 64; ++kk) h += g[gi * 64 + kk] * vw1[kk * 64 + d];
    acc += lrelu(h) * vw2[d];
  }
  out[gi] = acc;
}

// ---------------------------------------------------------------------------
// Host launcher
// ---------------------------------------------------------------------------
extern "C" void kernel_launch(void* const* d_in, const int* in_sizes, int n_in,
                              void* d_out, int out_size, void* d_ws,
                              size_t ws_size, hipStream_t stream) {
  (void)n_in; (void)out_size; (void)ws_size;
  const float* x         = (const float*)d_in[0];
  const float* edge_attr = (const float*)d_in[1];
  const float* u         = (const float*)d_in[2];
  const float* emb_e_w   = (const float*)d_in[3];
  const float* emb_e_b   = (const float*)d_in[4];
  const float* emb_n_w   = (const float*)d_in[5];
  const float* emb_n_b   = (const float*)d_in[6];
  const float* emb_g_w   = (const float*)d_in[7];
  const float* emb_g_b   = (const float*)d_in[8];
  const float* ew1 = (const float*)d_in[9];   const float* eb1 = (const float*)d_in[10];
  const float* ew2 = (const float*)d_in[11];  const float* eb2 = (const float*)d_in[12];
  const float* ew3 = (const float*)d_in[13];  const float* eb3 = (const float*)d_in[14];
  const float* nw1 = (const float*)d_in[15];  const float* nb1 = (const float*)d_in[16];
  const float* nw2 = (const float*)d_in[17];  const float* nb2 = (const float*)d_in[18];
  const float* nw3 = (const float*)d_in[19];  const float* nb3 = (const float*)d_in[20];
  const float* gw1 = (const float*)d_in[21];  const float* gb1 = (const float*)d_in[22];
  const float* gw2 = (const float*)d_in[23];  const float* gb2 = (const float*)d_in[24];
  const float* gw3 = (const float*)d_in[25];  const float* gb3 = (const float*)d_in[26];
  const float* dec_w = (const float*)d_in[27];
  const float* dec_b = (const float*)d_in[28];
  const float* vw1 = (const float*)d_in[29];  const float* vb1 = (const float*)d_in[30];
  const float* vw2 = (const float*)d_in[31];  const float* vb2 = (const float*)d_in[32];
  const int* edge_index = (const int*)d_in[33];
  const int* nbatch     = (const int*)d_in[34];

  const int N = in_sizes[0] / 16;
  const int E = in_sizes[1] / 16;
  const int B = in_sizes[2] / 16;
  const int P = in_sizes[10] / 128;  // eb1: [P,128]
  const int* erow = edge_index;
  const int* ecol = edge_index + E;

  char* ws = (char*)d_ws;
  size_t off = 0;
  auto alloc = [&](size_t bytes) -> void* {
    off = (off + 255) & ~(size_t)255;
    void* p = ws + off;
    off += bytes;
    return p;
  };
  float* xh    = (float*)alloc((size_t)N * 64 * 4);
  float* eF    = (float*)alloc((size_t)E * 64 * 4);
  float* gF    = (float*)alloc((size_t)B * 64 * 4);
  float* e_agg = (float*)alloc((size_t)N * 64 * 4);
  float* x_agg = (float*)alloc((size_t)B * 64 * 4);
  _Float16* ew1t = (_Float16*)alloc((size_t)P * 128 * 256 * 2);
  _Float16* ew2t = (_Float16*)alloc((size_t)P * 128 * 128 * 2);
  _Float16* ew3t = (_Float16*)alloc((size_t)P * 64 * 128 * 2);
  _Float16* nw1t = (_Float16*)alloc((size_t)P * 128 * 192 * 2);
  _Float16* nw2t = (_Float16*)alloc((size_t)P * 128 * 128 * 2);
  _Float16* nw3t = (_Float16*)alloc((size_t)P * 64 * 128 * 2);
  _Float16* gw1t = (_Float16*)alloc((size_t)P * 128 * 128 * 2);
  _Float16* gw2t = (_Float16*)alloc((size_t)P * 128 * 128 * 2);
  _Float16* gw3t = (_Float16*)alloc((size_t)P * 64 * 128 * 2);

  auto conv = [&](const float* in, _Float16* out, int K, int Nn) {
    int total = P * K * Nn;
    convert_w_kernel<<<(total + 255) / 256, 256, 0, stream>>>(in, out, P, K, Nn);
  };
  conv(ew1, ew1t, 256, 128); conv(ew2, ew2t, 128, 128); conv(ew3, ew3t, 128, 64);
  conv(nw1, nw1t, 192, 128); conv(nw2, nw2t, 128, 128); conv(nw3, nw3t, 128, 64);
  conv(gw1, gw1t, 128, 128); conv(gw2, gw2t, 128, 128); conv(gw3, gw3t, 128, 64);

  embed_kernel<<<((size_t)E * 64 + 255) / 256, 256, 0, stream>>>(edge_attr, emb_e_w, emb_e_b, eF, E);
  embed_kernel<<<((size_t)N * 64 + 255) / 256, 256, 0, stream>>>(x, emb_n_w, emb_n_b, xh, N);
  embed_kernel<<<((size_t)B * 64 + 255) / 256, 256, 0, stream>>>(u, emb_g_w, emb_g_b, gF, B);

  for (int p = 0; p < P; ++p) {
    hipMemsetAsync(e_agg, 0, (size_t)N * 64 * 4, stream);
    edge_mlp_kernel<<<(E + 63) / 64, 256, 0, stream>>>(
        xh, eF, gF, erow, ecol, nbatch,
        ew1t + (size_t)p * 128 * 256, eb1 + (size_t)p * 128,
        ew2t + (size_t)p * 128 * 128, eb2 + (size_t)p * 128,
        ew3t + (size_t)p * 64 * 128,  eb3 + (size_t)p * 64, e_agg, E);
    hipMemsetAsync(x_agg, 0, (size_t)B * 64 * 4, stream);
    node_mlp_kernel<<<(N + 63) / 64, 256, 0, stream>>>(
        xh, e_agg, gF, nbatch,
        nw1t + (size_t)p * 128 * 192, nb1 + (size_t)p * 128,
        nw2t + (size_t)p * 128 * 128, nb2 + (size_t)p * 128,
        nw3t + (size_t)p * 64 * 128,  nb3 + (size_t)p * 64, x_agg, N);
    global_mlp_kernel<<<1, 256, 0, stream>>>(
        gF, x_agg,
        gw1t + (size_t)p * 128 * 128, gb1 + (size_t)p * 128,
        gw2t + (size_t)p * 128 * 128, gb2 + (size_t)p * 128,
        gw3t + (size_t)p * 64 * 128,  gb3 + (size_t)p * 64, B);
  }

  float* out_f = (float*)d_out;
  edge_decode_kernel<<<(E + 7) / 8, 256, 0, stream>>>(eF, dec_w, dec_b, out_f, E);
  value_kernel<<<1, 64, 0, stream>>>(gF, vw1, vb1, vw2, vb2, out_f + E, B);
}
